// SchemaLinkingHomoGNN_15796889715042
// MI455X (gfx1250) — matrix operand order repaired and checked by reading, not compiled, not run on hardware
//
#include <hip/hip_runtime.h>
#include <hip/hip_bf16.h>

// ---------------- problem constants ----------------
#define N_NODES   100000
#define N_EDGES   1600000
#define IN_DIM    512
#define HID       64
#define N_LAYERS  3
#define TILES     (N_NODES / 16)       // 6250 16-node tiles, exact
#define BLOCK     256                  // 8 waves per block (wave32)

typedef __attribute__((ext_vector_type(16))) __bf16 v16bf;
typedef __attribute__((ext_vector_type(8)))  float  v8f;

// fp32 -> bf16 round-to-nearest-even
__device__ __forceinline__ __bf16 f2bf(float f) {
    unsigned int u = __builtin_bit_cast(unsigned int, f);
    unsigned int r = u + 0x7FFFu + ((u >> 16) & 1u);
    return __builtin_bit_cast(__bf16, (unsigned short)(r >> 16));
}

// ---------------- async global->LDS staging (CDNA5, ASYNCcnt-tracked) ----------------
// GVS mode: LDS address from the laddr VGPR; global address is the SGPR base plus the VGPR offset
__device__ __forceinline__ void lds_fill_issue(unsigned short* smem, const unsigned short* g, int nbytes) {
    unsigned lbase = (unsigned)(size_t)(void*)smem;   // low 32 bits = LDS offset
    for (int off = threadIdx.x * 16; off < nbytes; off += BLOCK * 16) {
        unsigned laddr = lbase + (unsigned)off;
        asm volatile("global_load_async_to_lds_b128 %0, %1, %2"
                     :: "v"(laddr), "v"(off), "s"(g)
                     : "memory");
    }
}
__device__ __forceinline__ void lds_fill_wait() {
    asm volatile("s_wait_asynccnt 0x0" ::: "memory");
    __syncthreads();
}

// ---------------- weight packing into WMMA B layout ----------------
// Packed element idx = ((kt*4 + nt)*32 + lane)*16 + j
// maps to W[kt*32 + (lane>>4)*8 + (j&7) + (j>>3)*16][nt*16 + (lane&15)]  (W is [K,64] row-major)
__global__ void pack_b_kernel(const float* __restrict__ src, unsigned short* __restrict__ dst, int K) {
    int i = blockIdx.x * blockDim.x + threadIdx.x;
    if (i >= K * HID) return;
    int j    = i & 15;
    int lane = (i >> 4) & 31;
    int nt   = (i >> 9) & 3;
    int kt   = i >> 11;
    int k    = kt * 32 + (lane >> 4) * 8 + (j & 7) + (j >> 3) * 16;
    int col  = nt * 16 + (lane & 15);
    unsigned int u = __builtin_bit_cast(unsigned int, src[(size_t)k * HID + col]);
    unsigned int r = u + 0x7FFFu + ((u >> 16) & 1u);
    dst[i] = (unsigned short)(r >> 16);
}

// ---------------- utility: zero fill ----------------
__global__ void zero_kernel(float* __restrict__ p, int n) {
    int i = blockIdx.x * blockDim.x + threadIdx.x;
    if (i < n) p[i] = 0.0f;
}

// ---------------- degree / inverse degree ----------------
__global__ void degree_kernel(const int* __restrict__ dst, float* __restrict__ deg) {
    int e = blockIdx.x * blockDim.x + threadIdx.x;
    if (e < N_EDGES) atomicAdd(&deg[dst[e]], 1.0f);
}
__global__ void invdeg_kernel(float* __restrict__ deg) {
    int i = blockIdx.x * blockDim.x + threadIdx.x;
    if (i < N_NODES) {
        float d = deg[i];
        deg[i] = (d > 0.0f) ? (1.0f / d) : 0.0f;
    }
}

// ---------------- edge scatter: agg[dst] += h[src]  (L2-resident atomics) ----------------
__global__ void scatter_kernel(const int* __restrict__ src, const int* __restrict__ dst,
                               const float* __restrict__ h, float* __restrict__ agg) {
    int t = blockIdx.x * blockDim.x + threadIdx.x;
    if (t >= N_EDGES * 16) return;
    int e = t >> 4;
    int g = (t & 15) * 4;
    int s = src[e], d = dst[e];
    const float4 v = *(const float4*)(h + (size_t)s * HID + g);
    float* ap = agg + (size_t)d * HID + g;
    atomicAdd(ap + 0, v.x);
    atomicAdd(ap + 1, v.y);
    atomicAdd(ap + 2, v.z);
    atomicAdd(ap + 3, v.w);
}

// ---------------- encoder: h = bf16(x) @ W_enc + b_enc ----------------
// Packed W_enc (64 KB) staged in LDS per block; one 16-node tile per wave.
__global__ void __launch_bounds__(BLOCK) encoder_wmma(const float* __restrict__ x,
                                                      const unsigned short* __restrict__ Wp,
                                                      const float* __restrict__ b_enc,
                                                      float* __restrict__ h) {
    __shared__ __align__(32) unsigned short sW[IN_DIM * HID];   // 32768 bf16 = 64 KB
    lds_fill_issue(sW, Wp, IN_DIM * HID * 2);
    lds_fill_wait();

    int wave = (int)((blockIdx.x * (unsigned)BLOCK + threadIdx.x) >> 5);
    int lane = threadIdx.x & 31;
    if (wave >= TILES) return;                       // wave-uniform guard (after barrier)
    int half = lane >> 4;
    int row  = wave * 16 + (lane & 15);              // A-matrix row for this lane
    const float* xr = x + (size_t)row * IN_DIM;
    const v16bf* Bs = (const v16bf*)sW;

    v8f acc0 = {}, acc1 = {}, acc2 = {}, acc3 = {};
    for (int kt = 0; kt < IN_DIM / 32; ++kt) {
        int k0 = kt * 32 + half * 8;
        v16bf a;
#pragma unroll
        for (int j = 0; j < 8; ++j) a[j] = f2bf(xr[k0 + j]);
#pragma unroll
        for (int j = 0; j < 8; ++j) a[8 + j] = f2bf(xr[k0 + 16 + j]);
        v16bf b0 = Bs[(kt * 4 + 0) * 32 + lane];
        v16bf b1 = Bs[(kt * 4 + 1) * 32 + lane];
        v16bf b2 = Bs[(kt * 4 + 2) * 32 + lane];
        v16bf b3 = Bs[(kt * 4 + 3) * 32 + lane];
        acc0 = __builtin_amdgcn_wmma_f32_16x16x32_bf16(false, a, false, b0, (short)0, acc0, false, false);
        acc1 = __builtin_amdgcn_wmma_f32_16x16x32_bf16(false, a, false, b1, (short)0, acc1, false, false);
        acc2 = __builtin_amdgcn_wmma_f32_16x16x32_bf16(false, a, false, b2, (short)0, acc2, false, false);
        acc3 = __builtin_amdgcn_wmma_f32_16x16x32_bf16(false, a, false, b3, (short)0, acc3, false, false);
    }
    int col = lane & 15;
#pragma unroll
    for (int v = 0; v < 8; ++v) {
        int m = wave * 16 + half * 8 + v;            // D-matrix row for VGPR v
        float* hr = h + (size_t)m * HID;
        hr[ 0 + col] = acc0[v] + b_enc[ 0 + col];
        hr[16 + col] = acc1[v] + b_enc[16 + col];
        hr[32 + col] = acc2[v] + b_enc[32 + col];
        hr[48 + col] = acc3[v] + b_enc[48 + col];
    }
}

// ---------------- fused SAGE layer: h = relu(h + (agg*invdeg)@Wl + bl + h@Wr) ----------------
__global__ void __launch_bounds__(BLOCK) layer_wmma(const float* __restrict__ agg,
                                                    const float* __restrict__ invdeg,
                                                    const unsigned short* __restrict__ Wlp,
                                                    const unsigned short* __restrict__ Wrp,
                                                    const float* __restrict__ bl,
                                                    float* __restrict__ h) {
    __shared__ __align__(32) unsigned short sWl[HID * HID];     // 4096 bf16 = 8 KB
    __shared__ __align__(32) unsigned short sWr[HID * HID];     // 4096 bf16 = 8 KB
    lds_fill_issue(sWl, Wlp, HID * HID * 2);
    lds_fill_issue(sWr, Wrp, HID * HID * 2);
    lds_fill_wait();

    int wave = (int)((blockIdx.x * (unsigned)BLOCK + threadIdx.x) >> 5);
    int lane = threadIdx.x & 31;
    if (wave >= TILES) return;
    int half = lane >> 4;
    int row  = wave * 16 + (lane & 15);
    const float* ar  = agg + (size_t)row * HID;
    const float* hr0 = h + (size_t)row * HID;
    float inv = invdeg[row];
    const v16bf* Bl = (const v16bf*)sWl;
    const v16bf* Br = (const v16bf*)sWr;

    v8f acc[4] = {};
#pragma unroll
    for (int kt = 0; kt < 2; ++kt) {                 // K = 64 -> 2 k-tiles
        int k0 = kt * 32 + half * 8;
        v16bf a1, a2;
#pragma unroll
        for (int j = 0; j < 8; ++j) {
            a1[j]     = f2bf(ar[k0 + j] * inv);
            a1[8 + j] = f2bf(ar[k0 + 16 + j] * inv);
            a2[j]     = f2bf(hr0[k0 + j]);
            a2[8 + j] = f2bf(hr0[k0 + 16 + j]);
        }
#pragma unroll
        for (int nt = 0; nt < 4; ++nt) {
            v16bf bL = Bl[(kt * 4 + nt) * 32 + lane];
            acc[nt] = __builtin_amdgcn_wmma_f32_16x16x32_bf16(false, a1, false, bL, (short)0, acc[nt], false, false);
            v16bf bR = Br[(kt * 4 + nt) * 32 + lane];
            acc[nt] = __builtin_amdgcn_wmma_f32_16x16x32_bf16(false, a2, false, bR, (short)0, acc[nt], false, false);
        }
    }
    int col = lane & 15;
#pragma unroll
    for (int v = 0; v < 8; ++v) {
        int m = wave * 16 + half * 8 + v;
        float* hrow = h + (size_t)m * HID;
#pragma unroll
        for (int nt = 0; nt < 4; ++nt) {
            int n = nt * 16 + col;
            float val = hrow[n] + acc[nt][v] + bl[n]; // residual + bias
            hrow[n] = val > 0.0f ? val : 0.0f;        // relu
        }
    }
}

// ---------------- classifier: out = sigmoid(h @ W_cls + b_cls) ----------------
__global__ void cls_kernel(const float* __restrict__ h, const float* __restrict__ Wc,
                           const float* __restrict__ bc, float* __restrict__ out) {
    int i = blockIdx.x * blockDim.x + threadIdx.x;
    if (i >= N_NODES) return;
    const float* hr = h + (size_t)i * HID;
    float s = bc[0];
#pragma unroll
    for (int j = 0; j < HID; ++j) s = fmaf(hr[j], Wc[j], s);
    out[i] = 1.0f / (1.0f + __expf(-s));
}

// ---------------- host side ----------------
extern "C" void kernel_launch(void* const* d_in, const int* in_sizes, int n_in,
                              void* d_out, int out_size, void* d_ws, size_t ws_size,
                              hipStream_t stream) {
    const float* x     = (const float*)d_in[0];
    const int*   eidx  = (const int*)d_in[1];     // [2, E]
    const float* W_enc = (const float*)d_in[2];
    const float* b_enc = (const float*)d_in[3];
    const float* Wl    = (const float*)d_in[4];   // [3,64,64]
    const float* bl    = (const float*)d_in[5];   // [3,64]
    const float* Wr    = (const float*)d_in[6];   // [3,64,64]
    const float* W_cls = (const float*)d_in[7];
    const float* b_cls = (const float*)d_in[8];
    float* out = (float*)d_out;

    const int* src = eidx;
    const int* dst = eidx + N_EDGES;

    // workspace layout (256B aligned)
    char* ws = (char*)d_ws;
    float* h    = (float*)(ws + 0);                       // 25,600,000 B
    float* agg  = (float*)(ws + 25600000);                // 25,600,000 B
    float* deg  = (float*)(ws + 51200000);                //    400,000 B (degree -> invdeg in place)
    unsigned short* packW = (unsigned short*)(ws + 51600128);
    unsigned short* packEnc = packW;                      // 32768 elems
    unsigned short* packWl  = packW + 32768;              // 3 * 4096
    unsigned short* packWr  = packW + 32768 + 3 * 4096;   // 3 * 4096

    const int T = BLOCK;

    // 1. pack weights into WMMA B layout (bf16)
    pack_b_kernel<<<(IN_DIM * HID + T - 1) / T, T, 0, stream>>>(W_enc, packEnc, IN_DIM);
    for (int l = 0; l < N_LAYERS; ++l) {
        pack_b_kernel<<<(HID * HID + T - 1) / T, T, 0, stream>>>(Wl + (size_t)l * HID * HID, packWl + l * 4096, HID);
        pack_b_kernel<<<(HID * HID + T - 1) / T, T, 0, stream>>>(Wr + (size_t)l * HID * HID, packWr + l * 4096, HID);
    }

    // 2. in-degree -> inverse degree
    zero_kernel<<<(N_NODES + T - 1) / T, T, 0, stream>>>(deg, N_NODES);
    degree_kernel<<<(N_EDGES + T - 1) / T, T, 0, stream>>>(dst, deg);
    invdeg_kernel<<<(N_NODES + T - 1) / T, T, 0, stream>>>(deg);

    // 3. encoder GEMM (WMMA): one 16-node tile per wave, 8 waves per block
    int enc_blocks = (TILES + 7) / 8;
    encoder_wmma<<<enc_blocks, T, 0, stream>>>(x, packEnc, b_enc, h);

    // 4. SAGE layers
    for (int l = 0; l < N_LAYERS; ++l) {
        zero_kernel<<<(N_NODES * HID + T - 1) / T, T, 0, stream>>>(agg, N_NODES * HID);
        scatter_kernel<<<(N_EDGES * 16 + T - 1) / T, T, 0, stream>>>(src, dst, h, agg);
        layer_wmma<<<enc_blocks, T, 0, stream>>>(agg, deg, packWl + l * 4096, packWr + l * 4096,
                                                 bl + (size_t)l * HID, h);
    }

    // 5. classifier
    cls_kernel<<<(N_NODES + T - 1) / T, T, 0, stream>>>(h, W_cls, b_cls, out);
}
